// MultiHeadAttention_2817498546410
// MI455X (gfx1250) — matrix operand order repaired and checked by reading, not compiled, not run on hardware
//
#include <hip/hip_runtime.h>

#define D_MODEL 1024
#define N_HEADS 16
#define D_KH    64
#define SEQ     2048
#define BATCH   2
#define M_TOT   (BATCH * SEQ)   // 4096 rows total

typedef __bf16 v16bf __attribute__((ext_vector_type(16)));
typedef float  v8f   __attribute__((ext_vector_type(8)));

// ---- CDNA5 async global->LDS copy (ASYNCcnt-tracked) ------------------------
// ISA 10.x / 15.18.3: GLOBAL_LOAD_ASYNC_TO_LDS_B128; VDST = per-lane LDS byte
// offset (flat LDS addr truncates to addr[31:0]), VADDR = 64-bit global addr.
__device__ __forceinline__ void async_ld_b128(const void* g, const void* lds) {
  asm volatile("global_load_async_to_lds_b128 %0, %1, off"
               :: "v"((uint32_t)(size_t)lds), "v"((uint64_t)(size_t)g)
               : "memory");
}
__device__ __forceinline__ void wait_async0() {
  asm volatile("s_wait_asynccnt 0x0" ::: "memory");
}

// ---- WMMA fragment helpers (ISA 7.12.2 layouts, wave32) ----------------------
// A-matrix 16x32 bf16: lane L holds row M=L%16, K = {kb..kb+7, kb+16..kb+23},
// kb = (L/16)*8  -> two contiguous 16B loads.
__device__ __forceinline__ v16bf frag_a(const __bf16* p0, const __bf16* p1) {
  union { uint4 u[2]; v16bf v; } f;
  f.u[0] = *(const uint4*)p0;
  f.u[1] = *(const uint4*)p1;
  return f.v;
}
// B-matrix 32x16 bf16: lane L holds column N=L%16, K = {kb..kb+7, kb+16..kb+23},
// kb = (L/16)*8  -> two contiguous 16B loads.
__device__ __forceinline__ v16bf frag_b(const __bf16* p) {
  union { uint4 u[2]; v16bf v; } f;
  f.u[0] = *(const uint4*)p;
  f.u[1] = *(const uint4*)(p + 16);
  return f.v;
}
__device__ __forceinline__ v8f wmma_bf16(v16bf a, v16bf b, v8f c) {
  // (neg_a, A, neg_b, B, c_mod, C, reuse_a, reuse_b)
  return __builtin_amdgcn_wmma_f32_16x16x32_bf16(false, a, false, b, (short)0, c,
                                                 false, false);
}

// ---- f32 -> bf16 conversion (vectorized) ------------------------------------
__global__ __launch_bounds__(256) void cvt_bf16_kernel(const float4* __restrict__ in,
                                                       __bf16* __restrict__ out,
                                                       int n4) {
  int i = blockIdx.x * 256 + threadIdx.x;
  if (i < n4) {
    float4 f = in[i];
    union { __bf16 b[4]; uint2 u; } r;
    r.b[0] = (__bf16)f.x; r.b[1] = (__bf16)f.y;
    r.b[2] = (__bf16)f.z; r.b[3] = (__bf16)f.w;
    ((uint2*)out)[i] = r.u;
  }
}

// ---- C = A[M,K] @ Bw[N,K]^T  (torch Linear, bias-free) ----------------------
// Block: 256 threads = 8 waves; tile 128x64; each wave computes 32x32.
// Double-buffered LDS staging via async global->LDS copies: one
// s_wait_asynccnt + one barrier per K-slice, WMMA stream stays busy.
template <bool OUT_F32>
__global__ __launch_bounds__(256) void gemm_xwt_kernel(
    const __bf16* __restrict__ A, const __bf16* __restrict__ Bw,
    float* __restrict__ Cf, __bf16* __restrict__ Cb, int M, int N, int K) {
  constexpr int BM = 128, BN = 64, BK = 32, LD = BK + 8;  // 40 halves = 80B rows
  __shared__ __bf16 At[2][BM][LD];
  __shared__ __bf16 Bt[2][BN][LD];
  const int tid = threadIdx.x, lane = tid & 31, wave = tid >> 5;
  const int wm = wave >> 1, wn = wave & 1;
  const int m16 = lane & 15, half = lane >> 4;
  const int kbA = half * 8, kbB = half * 8;
  const int row0 = blockIdx.y * BM, col0 = blockIdx.x * BN;
  const int ar = tid >> 1, ac = (tid & 1) * 16;  // A tile: 16 halves/thread
  const int br = tid >> 2, bc = (tid & 3) * 8;   // B tile:  8 halves/thread

  auto stage_tiles = [&](int k0, int buf) {
    const __bf16* ga = A + (size_t)(row0 + ar) * K + k0 + ac;
    async_ld_b128(ga,     &At[buf][ar][ac]);
    async_ld_b128(ga + 8, &At[buf][ar][ac + 8]);
    async_ld_b128(Bw + (size_t)(col0 + br) * K + k0 + bc, &Bt[buf][br][bc]);
  };

  v8f acc[2][2] = {};
  stage_tiles(0, 0);
  int buf = 0;
  for (int k0 = 0; k0 < K; k0 += BK) {
    wait_async0();      // this wave's copies into `buf` are done
    __syncthreads();    // everyone's copies into `buf` are done
    if (k0 + BK < K) stage_tiles(k0 + BK, buf ^ 1);
    v16bf af[2], bfr[2];
#pragma unroll
    for (int i = 0; i < 2; i++) {
      const __bf16* pr = &At[buf][wm * 32 + i * 16 + m16][0];
      af[i] = frag_a(pr + kbA, pr + kbA + 16);
    }
#pragma unroll
    for (int j = 0; j < 2; j++)
      bfr[j] = frag_b(&Bt[buf][wn * 32 + j * 16 + m16][kbB]);
#pragma unroll
    for (int i = 0; i < 2; i++)
#pragma unroll
      for (int j = 0; j < 2; j++)
        acc[i][j] = wmma_bf16(af[i], bfr[j], acc[i][j]);
    buf ^= 1;
  }
#pragma unroll
  for (int i = 0; i < 2; i++)
#pragma unroll
    for (int j = 0; j < 2; j++)
#pragma unroll
      for (int v = 0; v < 8; v++) {
        int r = row0 + wm * 32 + i * 16 + half * 8 + v;
        int c = col0 + wn * 32 + j * 16 + m16;
        if (OUT_F32) Cf[(size_t)r * N + c] = acc[i][j][v];
        else         Cb[(size_t)r * N + c] = (__bf16)acc[i][j][v];
      }
}

// ---- Causal flash attention: one block per (b, h, 64-query tile) ------------
// 4 waves; wave w owns query rows [w*16, w*16+16) x all 64 keys of a tile.
__global__ __launch_bounds__(128) void attn_kernel(const __bf16* __restrict__ Qp,
                                                   const __bf16* __restrict__ Kp,
                                                   const __bf16* __restrict__ Vp,
                                                   __bf16* __restrict__ ctx) {
  constexpr int LD = 72;  // 144B stride: 16B aligned, rotates LDS banks
  __shared__ __bf16 Kt[64][LD];        // [key][dim], filled by async copies
  __shared__ __bf16 Vt[64][LD];        // transposed: [dim][key]
  __shared__ __bf16 Pb[4][16][LD];     // per-wave P staging (C -> A layout)
  const int qt = blockIdx.x, h = blockIdx.y, b = blockIdx.z;
  const int tid = threadIdx.x, lane = tid & 31, wave = tid >> 5;
  const int m16 = lane & 15, half = lane >> 4;
  const int kbA = half * 8, kbB = half * 8;
  const float scale = 0.125f;  // 1/sqrt(d_k)

  // Q A-fragments for this wave's 16 rows (d_k = 64 -> two k-chunks of 32)
  const size_t qrow =
      (size_t)(b * SEQ + qt * 64 + wave * 16 + m16) * D_MODEL + h * D_KH;
  v16bf qa[2];
  qa[0] = frag_a(Qp + qrow + kbA,      Qp + qrow + kbA + 16);
  qa[1] = frag_a(Qp + qrow + 32 + kbA, Qp + qrow + 32 + kbA + 16);

  float mrow[8], lrow[8];
  v8f o[4] = {};
#pragma unroll
  for (int v = 0; v < 8; v++) { mrow[v] = -3.0e38f; lrow[v] = 0.0f; }

  const int r  = tid >> 1;          // staging: key row per thread pair
  const int c0 = (tid & 1) * 32;    // 32-halves column slab

  for (int kt = 0; kt <= qt; kt++) {   // causal: skip fully-masked key tiles
    __syncthreads();  // previous tile fully consumed before overwrite
    const size_t grow = (size_t)(b * SEQ + kt * 64 + r) * D_MODEL + h * D_KH + c0;
    // K tile: async global->LDS, row-major
#pragma unroll
    for (int i = 0; i < 4; i++)
      async_ld_b128(Kp + grow + 8 * i, &Kt[r][c0 + 8 * i]);
    // V tile: VGPR round trip with transpose (Vt[dim][key])
    {
      union { uint4 u[4]; __bf16 e[32]; } vv;
      const uint4* vg4 = (const uint4*)(Vp + grow);
#pragma unroll
      for (int i = 0; i < 4; i++) vv.u[i] = vg4[i];
#pragma unroll
      for (int i = 0; i < 32; i++) Vt[c0 + i][r] = vv.e[i];
    }
    if (kt < qt) {  // pull next tile toward the WGP while we compute
      __builtin_prefetch(Kp + grow + (size_t)64 * D_MODEL, 0, 3);
      __builtin_prefetch(Vp + grow + (size_t)64 * D_MODEL, 0, 3);
    }
    wait_async0();
    __syncthreads();

    // S = Q @ K^T  (4 n-tiles of 16; two WMMAs chained over d_k)
    v8f sc[4];
#pragma unroll
    for (int t = 0; t < 4; t++) {
      v8f c = {};
      c = wmma_bf16(qa[0], frag_b(&Kt[t * 16 + m16][kbB]),      c);
      c = wmma_bf16(qa[1], frag_b(&Kt[t * 16 + m16][32 + kbB]), c);
      sc[t] = c;
    }
#pragma unroll
    for (int t = 0; t < 4; t++)
#pragma unroll
      for (int v = 0; v < 8; v++) sc[t][v] *= scale;
    if (kt == qt) {  // diagonal tile: elementwise causal mask
#pragma unroll
      for (int t = 0; t < 4; t++)
#pragma unroll
        for (int v = 0; v < 8; v++) {
          int n = t * 16 + m16;                 // key within tile
          int m = wave * 16 + half * 8 + v;     // query within tile
          if (n > m) sc[t][v] = -1.0e30f;
        }
    }

    // Online softmax; each row lives in one 16-lane half -> shfl_xor over 16.
    float rs[8];
#pragma unroll
    for (int v = 0; v < 8; v++) {
      float mx = fmaxf(fmaxf(sc[0][v], sc[1][v]), fmaxf(sc[2][v], sc[3][v]));
      mx = fmaxf(mx, __shfl_xor(mx, 1));
      mx = fmaxf(mx, __shfl_xor(mx, 2));
      mx = fmaxf(mx, __shfl_xor(mx, 4));
      mx = fmaxf(mx, __shfl_xor(mx, 8));
      float mn = fmaxf(mrow[v], mx);
      float rr = __expf(mrow[v] - mn);
      float s = 0.0f;
#pragma unroll
      for (int t = 0; t < 4; t++) {
        float p = __expf(sc[t][v] - mn);
        sc[t][v] = p;
        s += p;
      }
      s += __shfl_xor(s, 1); s += __shfl_xor(s, 2);
      s += __shfl_xor(s, 4); s += __shfl_xor(s, 8);
      lrow[v] = lrow[v] * rr + s;
      mrow[v] = mn;
      rs[v] = rr;
    }
#pragma unroll
    for (int t = 0; t < 4; t++)
#pragma unroll
      for (int v = 0; v < 8; v++) o[t][v] *= rs[v];

    // C-layout P -> bf16 A-layout via per-wave LDS round trip
#pragma unroll
    for (int t = 0; t < 4; t++)
#pragma unroll
      for (int v = 0; v < 8; v++)
        Pb[wave][half * 8 + v][t * 16 + m16] = (__bf16)sc[t][v];
    __builtin_amdgcn_wave_barrier();  // order DS stores vs. DS loads in-wave
    v16bf pa0 = frag_a(&Pb[wave][m16][kbA],      &Pb[wave][m16][kbA + 16]);
    v16bf pa1 = frag_a(&Pb[wave][m16][32 + kbA], &Pb[wave][m16][32 + kbA + 16]);

    // O += P @ V  (V^T staged so B-fragments are contiguous)
#pragma unroll
    for (int t = 0; t < 4; t++) {
      o[t] = wmma_bf16(pa0, frag_b(&Vt[t * 16 + m16][kbB]),      o[t]);
      o[t] = wmma_bf16(pa1, frag_b(&Vt[t * 16 + m16][32 + kbB]), o[t]);
    }
  }

  float inv[8];
#pragma unroll
  for (int v = 0; v < 8; v++) inv[v] = 1.0f / lrow[v];
#pragma unroll
  for (int t = 0; t < 4; t++)
#pragma unroll
    for (int v = 0; v < 8; v++) {
      size_t orow = (size_t)(b * SEQ + qt * 64 + wave * 16 + half * 8 + v) * D_MODEL +
                    h * D_KH + t * 16 + m16;
      ctx[orow] = (__bf16)(o[t][v] * inv[v]);
    }
}

// ---- host orchestration ------------------------------------------------------
extern "C" void kernel_launch(void* const* d_in, const int* in_sizes, int n_in,
                              void* d_out, int out_size, void* d_ws, size_t ws_size,
                              hipStream_t stream) {
  (void)in_sizes; (void)n_in; (void)out_size; (void)ws_size;
  const float* q  = (const float*)d_in[0];
  const float* k  = (const float*)d_in[1];
  const float* v  = (const float*)d_in[2];
  // d_in[3] = mask (causal, known statically) -- unused
  const float* Wq = (const float*)d_in[4];
  const float* Wk = (const float*)d_in[5];
  const float* Wv = (const float*)d_in[6];
  const float* Wo = (const float*)d_in[7];
  float* out = (float*)d_out;

  char* ws = (char*)d_ws;
  size_t off = 0;
  auto alloc = [&](size_t bytes) -> __bf16* {
    __bf16* p = (__bf16*)(ws + off);
    off += (bytes + 255) & ~(size_t)255;
    return p;
  };
  const size_t XB = (size_t)M_TOT * D_MODEL * 2;     // 8 MiB each
  const size_t WB = (size_t)D_MODEL * D_MODEL * 2;   // 2 MiB each
  __bf16 *Xq = alloc(XB), *Xk = alloc(XB), *Xv = alloc(XB);
  __bf16 *wq = alloc(WB), *wk = alloc(WB), *wv = alloc(WB), *wo = alloc(WB);
  __bf16 *Qp = alloc(XB), *Kp = alloc(XB), *Vp = alloc(XB);
  __bf16 *cx = alloc(XB);                            // total = 64 MiB of d_ws

  auto cvt = [&](const float* src, __bf16* dst, int n) {
    int n4 = n / 4;
    cvt_bf16_kernel<<<(n4 + 255) / 256, 256, 0, stream>>>((const float4*)src, dst, n4);
  };
  cvt(q,  Xq, M_TOT * D_MODEL);
  cvt(k,  Xk, M_TOT * D_MODEL);
  cvt(v,  Xv, M_TOT * D_MODEL);
  cvt(Wq, wq, D_MODEL * D_MODEL);
  cvt(Wk, wk, D_MODEL * D_MODEL);
  cvt(Wv, wv, D_MODEL * D_MODEL);
  cvt(Wo, wo, D_MODEL * D_MODEL);

  dim3 ggrid(D_MODEL / 64, M_TOT / 128);
  gemm_xwt_kernel<false><<<ggrid, 256, 0, stream>>>(Xq, wq, nullptr, Qp,
                                                    M_TOT, D_MODEL, D_MODEL);
  gemm_xwt_kernel<false><<<ggrid, 256, 0, stream>>>(Xk, wk, nullptr, Kp,
                                                    M_TOT, D_MODEL, D_MODEL);
  gemm_xwt_kernel<false><<<ggrid, 256, 0, stream>>>(Xv, wv, nullptr, Vp,
                                                    M_TOT, D_MODEL, D_MODEL);

  attn_kernel<<<dim3(SEQ / 64, N_HEADS, BATCH), 128, 0, stream>>>(Qp, Kp, Vp, cx);

  gemm_xwt_kernel<true><<<ggrid, 256, 0, stream>>>(cx, wo, out, nullptr,
                                                   M_TOT, D_MODEL, D_MODEL);
}
